// GeneGraph_VIB_72146860638814
// MI455X (gfx1250) — matrix operand order repaired and checked by reading, not compiled, not run on hardware
//
#include <hip/hip_runtime.h>
#include <hip/hip_bf16.h>

// ---------------------------------------------------------------------------
// GeneGraph VIB forward for MI455X (gfx1250, wave32, WMMA).
// Heavy GEMMs (encoder, hp projection, att = hp @ hp^T, GCN) run on
// v_wmma_f32_16x16x32_bf16 with f32 accumulation. A-tile staging uses the
// CDNA5 async memory->LDS path (global_load_async_to_lds_b128 + ASYNCcnt).
// ---------------------------------------------------------------------------

#define TRY_ASYNC 1

typedef __attribute__((ext_vector_type(16))) __bf16 v16bf;
typedef __attribute__((ext_vector_type(8)))  float  v8f;

#define DEV __device__ __forceinline__

static constexpr int Bc  = 16;
static constexpr int Nc  = 978;
static constexpr int Ec  = 512;
static constexpr int EHc = 512;
static constexpr int IBc = 64;
static constexpr int DFc = 2304;
static constexpr int HIDc= 512;
static constexpr int Pc  = 8;
static constexpr int TOPKc = 50;
static constexpr int NEc = 50000;

DEV __bf16 f2bf(float f) {
  union { float f; unsigned u; } v; v.f = f;
  unsigned r = v.u + 0x7FFFu + ((v.u >> 16) & 1u);
  unsigned short h = (unsigned short)(r >> 16);
  __bf16 b; __builtin_memcpy(&b, &h, 2); return b;
}
DEV unsigned mapf(float f) {
  unsigned u = __float_as_uint(f);
  return (u >> 31) ? ~u : (u | 0x80000000u);
}
DEV float unmapf(unsigned m) {
  unsigned u = (m & 0x80000000u) ? (m & 0x7FFFFFFFu) : ~m;
  return __uint_as_float(u);
}
DEV float sigmoidf(float x) { return 1.0f / (1.0f + expf(-x)); }

// ---------------------------------------------------------------------------
// Generic bf16 WMMA GEMM:  C = act(alpha * op(A) @ op(B) + bias)
//   A: (M x K) row-major (or K x M if ATRANS -> A[m][k] = Ag[k*lda + m])
//   B: (K x N) row-major (or N x K if BTRANS -> computing A @ B^T)
//   C: f32 (Cf) and/or bf16 (Cb), both (M x N) with stride ldc
// Batched over blockIdx.z with  off = (z/bdiv)*s?o + (z%bdiv)*s?i
// Block: 256 threads (8 waves), tile 128x64, K-step 32.
// ---------------------------------------------------------------------------
union FragU { v16bf v; uint4 q[2]; };

template<int ACT, bool ATRANS, bool BTRANS>
__global__ __launch_bounds__(256)
void gemm_bf16_wmma(const __bf16* __restrict__ Ag, const __bf16* __restrict__ Bg,
                    const float* __restrict__ bias,
                    float* __restrict__ Cf, __bf16* __restrict__ Cb,
                    int M, int N, int K, int lda, int ldb, int ldc,
                    float alpha, int bdiv,
                    long long sAo, long long sAi,
                    long long sBo, long long sBi,
                    long long sCo, long long sCi)
{
  constexpr int BM = 128, BN = 64, BK = 32;
  constexpr int LDA_S = BK + 8;   // 40 elems -> 80B row stride (16B aligned)
  constexpr int LDB_S = BK + 8;
  __shared__ __align__(16) __bf16 As[BM * LDA_S];
  __shared__ __align__(16) __bf16 Bs[BN * LDB_S];

  const int z  = blockIdx.z;
  const int zo = z / bdiv, zi = z % bdiv;
  Ag += zo * sAo + zi * sAi;
  Bg += zo * sBo + zi * sBi;
  const long long coff = zo * sCo + zi * sCi;
  if (Cf) Cf += coff;
  if (Cb) Cb += coff;

  const int m0 = blockIdx.y * BM;
  const int n0 = blockIdx.x * BN;
  const int tid  = threadIdx.x;
  const int lane = tid & 31;
  const int wave = tid >> 5;
  const int wm = (wave >> 1) * 32;   // 0,32,64,96
  const int wn = (wave & 1) * 32;    // 0,32

  v8f acc00 = {}, acc01 = {}, acc10 = {}, acc11 = {};

  for (int k0 = 0; k0 < K; k0 += BK) {
    bool usedAsync = false;
    // ---------------- stage A tile: As[m][k] ----------------
    if (!ATRANS) {
      const int r = tid >> 1;
      const int c = (tid & 1) * 16;
      const int gm = m0 + r;
      if (gm < M && (k0 + c + 16) <= K && ((lda & 7) == 0)) {
        const __bf16* gsrc = Ag + (long long)gm * lda + (k0 + c);
        __bf16* sdst = &As[r * LDA_S + c];
#if TRY_ASYNC
        // CDNA5: async DMA memory -> LDS, tracked by ASYNCcnt (b128 = 16B/lane)
        {
          const unsigned lds0 = (unsigned)(unsigned long long)(sdst);
          const unsigned lds1 = (unsigned)(unsigned long long)(sdst + 8);
          const unsigned long long ga0 = (unsigned long long)(gsrc);
          const unsigned long long ga1 = (unsigned long long)(gsrc + 8);
          asm volatile("global_load_async_to_lds_b128 %0, %1, off"
                       :: "v"(lds0), "v"(ga0) : "memory");
          asm volatile("global_load_async_to_lds_b128 %0, %1, off"
                       :: "v"(lds1), "v"(ga1) : "memory");
          usedAsync = true;
        }
#else
        {
          const uint4* gp = reinterpret_cast<const uint4*>(gsrc);
          uint4* sp = reinterpret_cast<uint4*>(sdst);
          sp[0] = gp[0]; sp[1] = gp[1];
        }
#endif
      } else {
        // branch-free edge handling: clamp address, select-zero the value
        const int gmc = (gm < M) ? gm : (M - 1);
        #pragma unroll
        for (int i = 0; i < 16; ++i) {
          const int gk = k0 + c + i;
          const int gkc = (gk < K) ? gk : (K - 1);
          __bf16 v = Ag[(long long)gmc * lda + gkc];
          if (gm >= M || gk >= K) v = f2bf(0.0f);
          As[r * LDA_S + c + i] = v;
        }
      }
    } else {
      // A[m][k] = Ag[k*lda + m]; thread t: k = t/8, m-range (t%8)*16..+16
      const int kl = tid >> 3;
      const int mb = (tid & 7) * 16;
      const int gk = k0 + kl;
      const int gkc = (gk < K) ? gk : (K - 1);
      #pragma unroll
      for (int i = 0; i < 16; ++i) {
        const int gm = m0 + mb + i;
        const int gmc = (gm < M) ? gm : (M - 1);
        __bf16 v = Ag[(long long)gkc * lda + gmc];
        if (gm >= M || gk >= K) v = f2bf(0.0f);
        As[(mb + i) * LDA_S + kl] = v;
      }
    }
    // ---------------- stage B tile transposed: Bs[n][k] ----------------
    if (!BTRANS) {
      // Bg is (K x N); thread t: k = t/8, n-range (t%8)*8..+8
      const int kl = tid >> 3;
      const int nb = (tid & 7) * 8;
      const int gk = k0 + kl;
      const int gkc = (gk < K) ? gk : (K - 1);
      #pragma unroll
      for (int i = 0; i < 8; ++i) {
        const int gn = n0 + nb + i;
        const int gnc = (gn < N) ? gn : (N - 1);
        __bf16 v = Bg[(long long)gkc * ldb + gnc];
        if (gk >= K || gn >= N) v = f2bf(0.0f);
        Bs[(nb + i) * LDB_S + kl] = v;
      }
      if (k0 + BK < K)
        __builtin_prefetch(Bg + (long long)(gkc + BK) * ldb + n0 + nb, 0, 1);
    } else {
      // Bg is (N x K); thread t: n = t/4, k-part (t%4)*8
      const int nl = tid >> 2;
      const int kb = (tid & 3) * 8;
      const int gn = n0 + nl;
      if (gn < N && (k0 + kb + 8) <= K && ((ldb & 7) == 0)) {
        *reinterpret_cast<uint4*>(&Bs[nl * LDB_S + kb]) =
            *reinterpret_cast<const uint4*>(Bg + (long long)gn * ldb + k0 + kb);
        if (k0 + BK < K)
          __builtin_prefetch(Bg + (long long)gn * ldb + k0 + BK + kb, 0, 1);
      } else {
        const int gnc = (gn < N) ? gn : (N - 1);
        #pragma unroll
        for (int i = 0; i < 8; ++i) {
          const int gk = k0 + kb + i;
          const int gkc = (gk < K) ? gk : (K - 1);
          __bf16 v = Bg[(long long)gnc * ldb + gkc];
          if (gn >= N || gk >= K) v = f2bf(0.0f);
          Bs[nl * LDB_S + kb + i] = v;
        }
      }
    }
#if TRY_ASYNC
    if (usedAsync) asm volatile("s_wait_asynccnt 0x0" ::: "memory");
#endif
    (void)usedAsync;
    __syncthreads();

    // ---------------- fragments + WMMA ----------------
    const int half = lane >> 4;
    const int l15  = lane & 15;
    FragU a0, a1, b0, b1;
    {
      const __bf16* r0 = &As[(wm + l15) * LDA_S];
      a0.q[0] = *reinterpret_cast<const uint4*>(r0 + half * 8);
      a0.q[1] = *reinterpret_cast<const uint4*>(r0 + 16 + half * 8);
      const __bf16* r1 = &As[(wm + 16 + l15) * LDA_S];
      a1.q[0] = *reinterpret_cast<const uint4*>(r1 + half * 8);
      a1.q[1] = *reinterpret_cast<const uint4*>(r1 + 16 + half * 8);
      const __bf16* c0 = &Bs[(wn + l15) * LDB_S + half * 8];
      b0.q[0] = *reinterpret_cast<const uint4*>(c0);
      b0.q[1] = *reinterpret_cast<const uint4*>(c0 + 16);
      const __bf16* c1 = &Bs[(wn + 16 + l15) * LDB_S + half * 8];
      b1.q[0] = *reinterpret_cast<const uint4*>(c1);
      b1.q[1] = *reinterpret_cast<const uint4*>(c1 + 16);
    }
    acc00 = __builtin_amdgcn_wmma_f32_16x16x32_bf16(false, a0.v, false, b0.v, (short)0, acc00, false, false);
    acc01 = __builtin_amdgcn_wmma_f32_16x16x32_bf16(false, a0.v, false, b1.v, (short)0, acc01, false, false);
    acc10 = __builtin_amdgcn_wmma_f32_16x16x32_bf16(false, a1.v, false, b0.v, (short)0, acc10, false, false);
    acc11 = __builtin_amdgcn_wmma_f32_16x16x32_bf16(false, a1.v, false, b1.v, (short)0, acc11, false, false);
    __syncthreads();
  }

  // ---------------- epilogue ----------------
  const int l15  = lane & 15;
  const int half = lane >> 4;
  auto store_tile = [&](const v8f& acc, int tm, int tn) {
    const int col = tn + l15;
    if (col >= N) return;
    const float bz = bias ? bias[col] : 0.0f;
    #pragma unroll
    for (int v = 0; v < 8; ++v) {
      const int row = tm + v + 8 * half;
      if (row >= M) continue;
      float x = acc[v] * alpha + bz;
      if (ACT == 1)      x = fmaxf(x, 0.0f);
      else if (ACT == 2) x = 0.5f * x * (1.0f + erff(x * 0.70710678118f));
      const long long off = (long long)row * ldc + col;
      if (Cf) Cf[off] = x;
      if (Cb) Cb[off] = f2bf(x);
    }
  };
  store_tile(acc00, m0 + wm,      n0 + wn);
  store_tile(acc01, m0 + wm,      n0 + wn + 16);
  store_tile(acc10, m0 + wm + 16, n0 + wn);
  store_tile(acc11, m0 + wm + 16, n0 + wn + 16);
}

// ---------------------------------------------------------------------------
// Small VALU kernels
// ---------------------------------------------------------------------------
__global__ void k_f32_to_bf16(const float* __restrict__ s, __bf16* __restrict__ d, long long n) {
  long long i = (long long)blockIdx.x * 256 + threadIdx.x;
  if (i < n) d[i] = f2bf(s[i]);
}

__global__ void k_gate_emb(const float* __restrict__ x1, const float* __restrict__ gW,
                           const float* __restrict__ gb, const float* __restrict__ emb,
                           __bf16* __restrict__ h0) {
  long long idx = (long long)blockIdx.x * 256 + threadIdx.x;
  const long long tot = (long long)Bc * Nc * Ec;
  if (idx >= tot) return;
  const int e = (int)(idx % Ec);
  const long long bn = idx / Ec;
  const int n = (int)(bn % Nc);
  const int b = (int)(bn / Nc);
  const float g = sigmoidf(x1[b * Nc + n] * gW[e] + gb[e]);
  h0[idx] = f2bf(g * emb[(long long)n * Ec + e]);
}

__global__ void k_drug_l1(const float* __restrict__ d, const float* __restrict__ W1,
                          const float* __restrict__ b1, float* __restrict__ out) {
  const int idx = blockIdx.x * 256 + threadIdx.x;
  if (idx >= Bc * Ec) return;
  const int e = idx % Ec, b = idx / Ec;
  float acc = b1[e];
  const float* row = d + (long long)b * DFc;
  for (int k = 0; k < DFc; ++k) acc += row[k] * W1[(long long)k * Ec + e];
  out[idx] = fmaxf(acc, 0.0f);
}

__global__ void k_drug_l2_ln(const float* __restrict__ r1, const float* __restrict__ W2,
                             const float* __restrict__ b2, const float* __restrict__ lng,
                             const float* __restrict__ lnb, float* __restrict__ Dm) {
  const int b = blockIdx.x, di = blockIdx.y, j = threadIdx.x; // 128 threads
  const float* row = r1 + ((long long)di * Bc + b) * Ec;
  float t = b2[j];
  for (int k = 0; k < Ec; ++k) t += row[k] * W2[k * 128 + j];
  __shared__ float s1[128], s2[128];
  s1[j] = t; s2[j] = t * t; __syncthreads();
  for (int s = 64; s > 0; s >>= 1) {
    if (j < s) { s1[j] += s1[j + s]; s2[j] += s2[j + s]; }
    __syncthreads();
  }
  const float mu = s1[0] / 128.0f;
  const float var = s2[0] / 128.0f - mu * mu;
  Dm[((long long)di * Bc + b) * 128 + j] =
      (t - mu) * rsqrtf(var + 1e-5f) * lng[j] + lnb[j];
}

__global__ void k_feats(const float* __restrict__ Hf, const float* __restrict__ Dm,
                        const float* __restrict__ fm, __bf16* __restrict__ feats) {
  long long idx = (long long)blockIdx.x * 256 + threadIdx.x;
  const long long tot = (long long)Bc * Nc * 256;
  if (idx >= tot) return;
  const int d = (int)(idx % 256);
  const long long bn = idx / 256;
  const int b = (int)(bn / Nc);
  const float v = (d < 128) ? Hf[bn * 128 + d] : Dm[(long long)b * 128 + (d - 128)];
  feats[idx] = f2bf(v * sigmoidf(fm[d]));
}

__global__ void k_zero_f32(float* __restrict__ p, long long n) {
  long long i = (long long)blockIdx.x * 256 + threadIdx.x;
  if (i < n) p[i] = 0.0f;
}

__global__ void k_scatter_adj(const int* __restrict__ ei, float* __restrict__ adj) {
  const int i = blockIdx.x * 256 + threadIdx.x;
  if (i >= NEc) return;
  const int r = ei[i], c = ei[NEc + i];
  adj[(long long)r * Nc + c] = 1.0f;
}

// one block per attention row: top-k threshold (binary search on monotone
// float->uint map), softmax over selected, then 0.2*init + 0.8*softmax.
__global__ __launch_bounds__(256)
void k_topk_softmax_mix(float* __restrict__ att, const float* __restrict__ initAdj) {
  const int b = blockIdx.y, n = blockIdx.x, tid = threadIdx.x;
  float* row = att + ((long long)b * Nc + n) * Nc;
  __shared__ float s_row[Nc];
  __shared__ unsigned s_cnt, s_maxu;
  __shared__ float s_sum;
  for (int i = tid; i < Nc; i += 256) s_row[i] = row[i];
  if (tid == 0) { s_maxu = 0u; s_sum = 0.0f; }
  __syncthreads();
  unsigned lm = 0u;
  for (int i = tid; i < Nc; i += 256) { unsigned m = mapf(s_row[i]); lm = (m > lm) ? m : lm; }
  atomicMax(&s_maxu, lm);
  __syncthreads();
  const float vmax = unmapf(s_maxu);
  unsigned lo = 0u, hi = s_maxu;
  while (lo < hi) {
    const unsigned mid = lo + ((hi - lo + 1u) >> 1);
    if (tid == 0) s_cnt = 0u;
    __syncthreads();
    unsigned c = 0u;
    for (int i = tid; i < Nc; i += 256) if (mapf(s_row[i]) >= mid) c++;
    atomicAdd(&s_cnt, c);
    __syncthreads();
    if (s_cnt >= (unsigned)TOPKc) lo = mid; else hi = mid - 1u;
    __syncthreads();
  }
  const unsigned T = lo;
  float ls = 0.0f;
  for (int i = tid; i < Nc; i += 256)
    if (mapf(s_row[i]) >= T) ls += expf(s_row[i] - vmax);
  atomicAdd(&s_sum, ls);
  __syncthreads();
  const float inv = 1.0f / s_sum;
  const float* ia = initAdj + (long long)n * Nc;
  for (int i = tid; i < Nc; i += 256) {
    const float p = (mapf(s_row[i]) >= T) ? expf(s_row[i] - vmax) * inv : 0.0f;
    row[i] = 0.2f * ia[i] + 0.8f * p;
  }
}

__global__ void k_colsum_dinv(const float* __restrict__ adj, float* __restrict__ dinv) {
  const int id = blockIdx.x * 256 + threadIdx.x;
  if (id >= Bc * Nc) return;
  const int b = id / Nc, i = id % Nc;
  float s = 1.0f; // + eye
  const float* base = adj + (long long)b * Nc * Nc + i;
  for (int j = 0; j < Nc; ++j) s += base[(long long)j * Nc];
  dinv[id] = rsqrtf(s);
}

__global__ void k_norm_bf16(const float* __restrict__ adj, const float* __restrict__ dinv,
                            __bf16* __restrict__ nb) {
  long long idx = (long long)blockIdx.x * 256 + threadIdx.x;
  const long long tot = (long long)Bc * Nc * Nc;
  if (idx >= tot) return;
  const int i = (int)(idx % Nc);
  const long long bj = idx / Nc;
  const int j = (int)(bj % Nc);
  const int b = (int)(bj / Nc);
  const float a = adj[idx] + ((i == j) ? 1.0f : 0.0f);
  nb[idx] = f2bf(a * dinv[b * Nc + i] * dinv[b * Nc + j]);
}

__global__ void k_gemb_z(const float* __restrict__ node, const float* __restrict__ eps,
                         float* __restrict__ muO, float* __restrict__ stdO,
                         float* __restrict__ z) {
  const int b = blockIdx.x, f = threadIdx.x; // 128 threads
  float s = 0.0f;
  const float* base = node + (long long)b * Nc * 128 + f;
  for (int n = 0; n < Nc; ++n) s += base[(long long)n * 128];
  __shared__ float sg[128];
  sg[f] = s / (float)Nc;
  __syncthreads();
  if (f < IBc) {
    const float mu = sg[f];
    const float x = sg[IBc + f] - (float)IBc;
    const float sd = (x > 20.0f) ? x : log1pf(expf(x));
    muO[b * IBc + f] = mu;
    stdO[b * IBc + f] = sd;
    z[b * IBc + f] = mu + eps[b * IBc + f] * sd;
  }
}

__global__ void k_dec1(const float* __restrict__ z, const float* __restrict__ W,
                       const float* __restrict__ bb, float* __restrict__ d0) {
  const int idx = blockIdx.x * 256 + threadIdx.x;
  if (idx >= Bc * EHc) return;
  const int e = idx % EHc, b = idx / EHc;
  float acc = bb[e];
  for (int k = 0; k < IBc; ++k) acc += z[b * IBc + k] * W[k * EHc + e];
  d0[idx] = acc;
}

__global__ void k_bn_relu(const float* __restrict__ d0, const float* __restrict__ g,
                          const float* __restrict__ bb, float* __restrict__ dn) {
  const int e = blockIdx.x * 256 + threadIdx.x;
  if (e >= EHc) return;
  float vals[Bc];
  float s = 0.0f, s2 = 0.0f;
  for (int b = 0; b < Bc; ++b) {
    const float v = d0[b * EHc + e];
    vals[b] = v; s += v; s2 += v * v;
  }
  const float m = s / (float)Bc;
  const float var = s2 / (float)Bc - m * m;
  const float rs = rsqrtf(var + 1e-5f);
  for (int b = 0; b < Bc; ++b)
    dn[b * EHc + e] = fmaxf((vals[b] - m) * rs * g[e] + bb[e], 0.0f);
}

__global__ void k_dec2(const float* __restrict__ dn, const float* __restrict__ W,
                       const float* __restrict__ bb, float* __restrict__ out) {
  const int idx = blockIdx.x * 256 + threadIdx.x;
  if (idx >= Bc * Nc) return;
  const int n = idx % Nc, b = idx / Nc;
  float acc = bb[n];
  for (int k = 0; k < EHc; ++k) acc += dn[b * EHc + k] * W[(long long)k * Nc + n];
  out[idx] = fmaxf(acc, 0.0f);
}

// ---------------------------------------------------------------------------
// Launch
// ---------------------------------------------------------------------------
static inline size_t alignUp(size_t x) { return (x + 255) & ~(size_t)255; }
static inline int cdiv(long long a, int b) { return (int)((a + b - 1) / b); }

extern "C" void kernel_launch(void* const* d_in, const int* in_sizes, int n_in,
                              void* d_out, int out_size, void* d_ws, size_t ws_size,
                              hipStream_t stream) {
  (void)in_sizes; (void)n_in; (void)out_size; (void)ws_size;
  // inputs (setup_inputs order)
  const float* x1     = (const float*)d_in[0];
  const float* drug1  = (const float*)d_in[1];
  const float* drug2  = (const float*)d_in[2];
  const float* eps    = (const float*)d_in[3];
  const int*   edge   = (const int*)d_in[4];
  const float* gateW  = (const float*)d_in[5];
  const float* gateb  = (const float*)d_in[6];
  const float* emb    = (const float*)d_in[7];
  const float* encW1  = (const float*)d_in[8];
  const float* encb1  = (const float*)d_in[9];
  const float* encW2  = (const float*)d_in[10];
  const float* encb2  = (const float*)d_in[11];
  const float* netW1  = (const float*)d_in[12];
  const float* netb1  = (const float*)d_in[13];
  const float* netW2  = (const float*)d_in[14];
  const float* netb2  = (const float*)d_in[15];
  const float* ln_g   = (const float*)d_in[16];
  const float* ln_b   = (const float*)d_in[17];
  const float* fmask  = (const float*)d_in[18];
  const float* Wsims  = (const float*)d_in[19];
  const float* gcnW1  = (const float*)d_in[20];
  const float* gcnb1  = (const float*)d_in[21];
  const float* gcnW2  = (const float*)d_in[22];
  const float* gcnb2  = (const float*)d_in[23];
  const float* decW1  = (const float*)d_in[24];
  const float* decb1  = (const float*)d_in[25];
  const float* bn_g   = (const float*)d_in[26];
  const float* bn_b   = (const float*)d_in[27];
  const float* decW2  = (const float*)d_in[28];
  const float* decb2  = (const float*)d_in[29];

  // outputs: x2_pred (B*N), adj2 (B*N*N), mu (B*IB), std (B*IB)
  float* out    = (float*)d_out;
  float* x2O    = out;
  float* adj2O  = out + (long long)Bc * Nc;
  float* muO    = adj2O + (long long)Bc * Nc * Nc;
  float* stdO   = muO + (long long)Bc * IBc;

  // workspace carve-up
  char* p = (char*)d_ws;
  auto alloc = [&](size_t bytes) { char* r = p; p += alignUp(bytes); return r; };
  __bf16* encW1b = (__bf16*)alloc((size_t)Ec * EHc * 2);
  __bf16* encW2b = (__bf16*)alloc((size_t)EHc * 128 * 2);
  __bf16* WsimsB = (__bf16*)alloc((size_t)Pc * 256 * HIDc * 2);
  __bf16* gcnW1b = (__bf16*)alloc((size_t)128 * HIDc * 2);
  __bf16* gcnW2b = (__bf16*)alloc((size_t)HIDc * 128 * 2);
  __bf16* h0b    = (__bf16*)alloc((size_t)Bc * Nc * Ec * 2);
  __bf16* h1b    = (__bf16*)alloc((size_t)Bc * Nc * EHc * 2);
  float*  Hf     = (float*) alloc((size_t)Bc * Nc * 128 * 4);
  __bf16* Hb     = (__bf16*)alloc((size_t)Bc * Nc * 128 * 2);
  float*  r1     = (float*) alloc((size_t)2 * Bc * Ec * 4);
  float*  Dm     = (float*) alloc((size_t)2 * Bc * 128 * 4);
  __bf16* featsb = (__bf16*)alloc((size_t)Bc * Nc * 256 * 2);
  __bf16* hpb    = (__bf16*)alloc((size_t)Bc * Nc * Pc * HIDc * 2);
  float*  attw   = (float*) alloc((size_t)Bc * Nc * Nc * 4);   // adj1 after topk
  float*  initAdj= (float*) alloc((size_t)Nc * Nc * 4);
  float*  dinv   = (float*) alloc((size_t)Bc * Nc * 4);
  __bf16* normb  = (__bf16*)alloc((size_t)Bc * Nc * Nc * 2);
  __bf16* xWb    = (__bf16*)alloc((size_t)Bc * Nc * HIDc * 2);
  __bf16* tb     = (__bf16*)alloc((size_t)Bc * Nc * HIDc * 2);
  __bf16* xW2b   = (__bf16*)alloc((size_t)Bc * Nc * 128 * 2);
  float*  nodef  = (float*) alloc((size_t)Bc * Nc * 128 * 4);
  float*  zbuf   = (float*) alloc((size_t)Bc * IBc * 4);
  float*  d0buf  = (float*) alloc((size_t)Bc * EHc * 4);
  float*  dnbuf  = (float*) alloc((size_t)Bc * EHc * 4);

  const int MBN = Bc * Nc; // 15648
  auto g2 = [](int M, int N, int Z) { return dim3((unsigned)((N + 63) / 64), (unsigned)((M + 127) / 128), (unsigned)Z); };

  // --- weight conversions to bf16 ---
  k_f32_to_bf16<<<cdiv((long long)Ec * EHc, 256), 256, 0, stream>>>(encW1, encW1b, (long long)Ec * EHc);
  k_f32_to_bf16<<<cdiv((long long)EHc * 128, 256), 256, 0, stream>>>(encW2, encW2b, (long long)EHc * 128);
  k_f32_to_bf16<<<cdiv((long long)Pc * 256 * HIDc, 256), 256, 0, stream>>>(Wsims, WsimsB, (long long)Pc * 256 * HIDc);
  k_f32_to_bf16<<<cdiv((long long)128 * HIDc, 256), 256, 0, stream>>>(gcnW1, gcnW1b, (long long)128 * HIDc);
  k_f32_to_bf16<<<cdiv((long long)HIDc * 128, 256), 256, 0, stream>>>(gcnW2, gcnW2b, (long long)HIDc * 128);

  // --- encoder ---
  k_gate_emb<<<cdiv((long long)Bc * Nc * Ec, 256), 256, 0, stream>>>(x1, gateW, gateb, emb, h0b);
  gemm_bf16_wmma<2, false, false><<<g2(MBN, EHc, 1), 256, 0, stream>>>(
      h0b, encW1b, encb1, nullptr, h1b, MBN, EHc, Ec, Ec, EHc, EHc,
      1.0f, 1, 0, 0, 0, 0, 0, 0);
  gemm_bf16_wmma<0, false, false><<<g2(MBN, 128, 1), 256, 0, stream>>>(
      h1b, encW2b, encb2, Hf, Hb, MBN, 128, EHc, EHc, 128, 128,
      1.0f, 1, 0, 0, 0, 0, 0, 0);

  // --- drug embeddings ---
  k_drug_l1<<<cdiv(Bc * Ec, 256), 256, 0, stream>>>(drug1, netW1, netb1, r1);
  k_drug_l1<<<cdiv(Bc * Ec, 256), 256, 0, stream>>>(drug2, netW1, netb1, r1 + (long long)Bc * Ec);
  k_drug_l2_ln<<<dim3(Bc, 2), 128, 0, stream>>>(r1, netW2, netb2, ln_g, ln_b, Dm);

  // --- init adjacency ---
  k_zero_f32<<<cdiv((long long)Nc * Nc, 256), 256, 0, stream>>>(initAdj, (long long)Nc * Nc);
  k_scatter_adj<<<cdiv(NEc, 256), 256, 0, stream>>>(edge, initAdj);

  // --- learned adjacency: fi=0 -> adj1 (ws), fi=1 -> adj2 (d_out) ---
  for (int fi = 0; fi < 2; ++fi) {
    float* attDst = (fi == 0) ? attw : adj2O;
    k_feats<<<cdiv((long long)Bc * Nc * 256, 256), 256, 0, stream>>>(
        Hf, Dm + (long long)fi * Bc * 128, fmask, featsb);
    // hp[b,n,p*512+h] = relu(feats[b] @ Wsims[p]); batched z = b*8+p
    gemm_bf16_wmma<1, false, false><<<g2(Nc, HIDc, Bc * Pc), 256, 0, stream>>>(
        featsb, WsimsB, nullptr, nullptr, hpb, Nc, HIDc, 256, 256, HIDc, Pc * HIDc,
        1.0f, Pc,
        (long long)Nc * 256, 0,                 // A: per-b
        0, (long long)256 * HIDc,               // B: per-p
        (long long)Nc * Pc * HIDc, (long long)HIDc); // C: b-major, p-offset
    // att[b] = (hp[b] @ hp[b]^T) / P
    gemm_bf16_wmma<0, false, true><<<g2(Nc, Nc, Bc), 256, 0, stream>>>(
        hpb, hpb, nullptr, attDst, nullptr, Nc, Nc, Pc * HIDc,
        Pc * HIDc, Pc * HIDc, Nc,
        1.0f / (float)Pc, 1,
        (long long)Nc * Pc * HIDc, 0,
        (long long)Nc * Pc * HIDc, 0,
        (long long)Nc * Nc, 0);
    k_topk_softmax_mix<<<dim3(Nc, Bc), 256, 0, stream>>>(attDst, initAdj);
  }

  // --- GCN with adj1 ---
  k_colsum_dinv<<<cdiv(Bc * Nc, 256), 256, 0, stream>>>(attw, dinv);
  k_norm_bf16<<<cdiv((long long)Bc * Nc * Nc, 256), 256, 0, stream>>>(attw, dinv, normb);
  // xW = H @ gcnW1
  gemm_bf16_wmma<0, false, false><<<g2(MBN, HIDc, 1), 256, 0, stream>>>(
      Hb, gcnW1b, nullptr, nullptr, xWb, MBN, HIDc, 128, 128, HIDc, HIDc,
      1.0f, 1, 0, 0, 0, 0, 0, 0);
  // t = relu(norm^T @ xW + b1), per b
  gemm_bf16_wmma<1, true, false><<<g2(Nc, HIDc, Bc), 256, 0, stream>>>(
      normb, xWb, gcnb1, nullptr, tb, Nc, HIDc, Nc, Nc, HIDc, HIDc,
      1.0f, 1,
      (long long)Nc * Nc, 0, (long long)Nc * HIDc, 0, (long long)Nc * HIDc, 0);
  // xW2 = t @ gcnW2
  gemm_bf16_wmma<0, false, false><<<g2(MBN, 128, 1), 256, 0, stream>>>(
      tb, gcnW2b, nullptr, nullptr, xW2b, MBN, 128, HIDc, HIDc, 128, 128,
      1.0f, 1, 0, 0, 0, 0, 0, 0);
  // node = norm^T @ xW2 + b2, per b (f32 out)
  gemm_bf16_wmma<0, true, false><<<g2(Nc, 128, Bc), 256, 0, stream>>>(
      normb, xW2b, gcnb2, nodef, nullptr, Nc, 128, Nc, Nc, 128, 128,
      1.0f, 1,
      (long long)Nc * Nc, 0, (long long)Nc * 128, 0, (long long)Nc * 128, 0);

  // --- VAE head + decoder ---
  k_gemb_z<<<Bc, 128, 0, stream>>>(nodef, eps, muO, stdO, zbuf);
  k_dec1<<<cdiv(Bc * EHc, 256), 256, 0, stream>>>(zbuf, decW1, decb1, d0buf);
  k_bn_relu<<<cdiv(EHc, 256), 256, 0, stream>>>(d0buf, bn_g, bn_b, dnbuf);
  k_dec2<<<cdiv(Bc * Nc, 256), 256, 0, stream>>>(dnbuf, decW2, decb2, x2O);
}
